// CausalSelfAttention_32186484916684
// MI455X (gfx1250) — matrix operand order repaired and checked by reading, not compiled, not run on hardware
//
#include <hip/hip_runtime.h>

typedef __attribute__((ext_vector_type(16))) _Float16 v16h;
typedef __attribute__((ext_vector_type(8)))  _Float16 v8h;
typedef __attribute__((ext_vector_type(4)))  _Float16 v4h;
typedef __attribute__((ext_vector_type(8)))  float    v8f;

constexpr int Bb    = 2;
constexpr int Ss    = 2048;
constexpr int Dd    = 1024;
constexpr int Hh    = 16;
constexpr int HDd   = 64;
constexpr int Mrows = Bb * Ss;          // 4096
constexpr float SCALE = 0.125f;         // hd^-0.5

// padded LDS row strides (halves): bank-conflict-free for 16-lane row reads
constexpr int KPAD = 72;                // 32 K rows  x 64 data halves
constexpr int VPAD = 40;                // 64 V rows  x 32 data halves
constexpr int WPAD = 40;                // 64 W rows  x 32 data halves

// ---------------------------------------------------------------------------
// WMMA operand loaders (layouts per CDNA5 ISA 7.12.2, wave32)
// A (16x32 f16): lane L -> row m=L&15; halves 0..7 = K[kb..kb+8),
//                halves 8..15 = K[kb+16..kb+24), kb = 8*(L>>4)
// B (32x16 f16): lane L -> col n=L&15; same K order as A
// C/D (16x16 f32): vgpr r -> row r + 8*(L>>4); lane -> col L&15
// ---------------------------------------------------------------------------
__device__ inline v16h load_a_tile(const _Float16* base, int stride, int k0, int lane) {
    const int m  = lane & 15;
    const int kb = (lane >> 4) * 8;
    const _Float16* p = base + (size_t)m * stride + k0 + kb;
    union { v16h v; v8h h[2]; } u;
    u.h[0] = *(const v8h*)(p);
    u.h[1] = *(const v8h*)(p + 16);
    return u.v;
}

__device__ inline v16h load_b_tile(const _Float16* base, int stride, int k0, int lane) {
    const _Float16* p = base + (size_t)(lane & 15) * stride + k0 + (lane >> 4) * 8;
    union { v16h v; v8h h[2]; } u;
    u.h[0] = *(const v8h*)(p);
    u.h[1] = *(const v8h*)(p + 16);
    return u.v;
}

__device__ inline v8f wmma_f16(v16h a, v16h b, v8f c) {
    return __builtin_amdgcn_wmma_f32_16x16x32_f16(false, a, false, b, (short)0, c, false, false);
}

// gfx1250 async copy: global -> LDS, 16B per lane, tracked by ASYNCcnt
__device__ inline void async_b128(unsigned lds_off, const _Float16* gptr) {
    asm volatile("global_load_async_to_lds_b128 %0, %1, off"
                 :: "v"(lds_off), "v"((unsigned long long)(size_t)gptr)
                 : "memory");
}

__device__ inline float rowmax16(float v) {
    v = fmaxf(v, __shfl_xor(v, 1, 32));
    v = fmaxf(v, __shfl_xor(v, 2, 32));
    v = fmaxf(v, __shfl_xor(v, 4, 32));
    v = fmaxf(v, __shfl_xor(v, 8, 32));
    return v;
}
__device__ inline float rowsum16(float v) {
    v += __shfl_xor(v, 1, 32);
    v += __shfl_xor(v, 2, 32);
    v += __shfl_xor(v, 4, 32);
    v += __shfl_xor(v, 8, 32);
    return v;
}

// ---------------------------------------------------------------------------
// fp32 -> fp16 conversion
// ---------------------------------------------------------------------------
__global__ void cvt_f32_f16_kernel(const float* __restrict__ src,
                                   _Float16* __restrict__ dst, int n4) {
    int i      = blockIdx.x * blockDim.x + threadIdx.x;
    int stride = gridDim.x * blockDim.x;
    for (; i < n4; i += stride) {
        float4 f = ((const float4*)src)[i];
        v4h h;
        h[0] = (_Float16)f.x; h[1] = (_Float16)f.y;
        h[2] = (_Float16)f.z; h[3] = (_Float16)f.w;
        ((v4h*)dst)[i] = h;
    }
}

// ---------------------------------------------------------------------------
// GEMM: C[4096,1024] = A[4096,1024] * W[1024,1024]^T  (f16 in, f32 accum)
// 8 waves/WG, wave -> 16x64 tile; WG shares the 64x32 W tile via async LDS
// staging, double-buffered. grid = (32, 16).
// mode 0: RoPE + store f16 [B,H,S,64]   (Q and K)
// mode 1: store f16 transposed [B,H,64,S] (Vt)
// mode 2: store f32 [M,N] (output projection)
// ---------------------------------------------------------------------------
__global__ void gemm_kernel(const _Float16* __restrict__ A,
                            const _Float16* __restrict__ W,
                            void* __restrict__ out, int mode) {
    __shared__ __align__(16) _Float16 Wbuf[2][64 * WPAD];

    const int tid  = threadIdx.x;
    const int lane = tid & 31;
    const int wave = tid >> 5;
    const int m0   = blockIdx.x * 128 + wave * 16;
    const int n0   = blockIdx.y * 64;

    // staging role: 64 rows x 4 chunks of 16B
    const int wrow   = tid >> 2;
    const int wchunk = tid & 3;
    const _Float16* wgp = W + (size_t)(n0 + wrow) * Dd + wchunk * 8;
    const unsigned wlds[2] = {
        (unsigned)(size_t)&Wbuf[0][wrow * WPAD + wchunk * 8],
        (unsigned)(size_t)&Wbuf[1][wrow * WPAD + wchunk * 8] };

    v8f acc[4] = {};
    const _Float16* Abase = A + (size_t)m0 * Dd;

    constexpr int KT = Dd / 32;                       // 32 k-steps
    async_b128(wlds[0], wgp);                         // prologue: tile 0

    for (int kt = 0; kt < KT; ++kt) {
        if (kt + 1 < KT) {
            async_b128(wlds[(kt + 1) & 1], wgp + (kt + 1) * 32);
            asm volatile("s_wait_asynccnt 0x1" ::: "memory");
        } else {
            asm volatile("s_wait_asynccnt 0x0" ::: "memory");
        }
        __syncthreads();                              // tile kt visible to all

        v16h a = load_a_tile(Abase, Dd, kt * 32, lane);
        const _Float16* wb = &Wbuf[kt & 1][0];
#pragma unroll
        for (int ct = 0; ct < 4; ++ct) {
            v16h b = load_b_tile(wb + ct * 16 * WPAD, WPAD, 0, lane);
            acc[ct] = wmma_f16(a, b, acc[ct]);
        }
        __syncthreads();                              // reads done before reuse
    }

    const int nloc  = lane & 15;
    const int rbase = (lane >> 4) * 8;
    const int s0    = m0 & (Ss - 1);
    const int bidx  = m0 >> 11;
    const int h     = n0 >> 6;

    if (mode == 0) {                    // Q/K: RoPE then [B,H,S,64] f16
        _Float16* O = (_Float16*)out;
#pragma unroll
        for (int ct = 0; ct < 4; ++ct) {
            const int d = ct * 16 + nloc;
            const float invf = __expf(-(float)(d & ~1) * (9.210340371976184f / 64.0f));
#pragma unroll
            for (int r = 0; r < 8; ++r) {
                const int s = s0 + rbase + r;
                const float theta = (float)s * invf;
                const float cs = cosf(theta), sn = sinf(theta);
                const float val  = acc[ct][r];
                const float part = __shfl_xor(val, 1, 32);
                const float res  = (d & 1) ? (val * cs + part * sn)
                                           : (val * cs - part * sn);
                O[((size_t)(bidx * Hh + h) * Ss + s) * HDd + d] = (_Float16)res;
            }
        }
    } else if (mode == 1) {             // V: store transposed [B,H,64,S]
        _Float16* Vt = (_Float16*)out;
        const int sb = s0 + rbase;
#pragma unroll
        for (int ct = 0; ct < 4; ++ct) {
            const int d = ct * 16 + nloc;
            v8h pack;
#pragma unroll
            for (int r = 0; r < 8; ++r) pack[r] = (_Float16)acc[ct][r];
            *(v8h*)(Vt + ((size_t)((bidx * Hh + h) * HDd + d)) * Ss + sb) = pack;
        }
    } else {                            // output projection: f32 [M,N]
        float* O = (float*)out;
#pragma unroll
        for (int ct = 0; ct < 4; ++ct)
#pragma unroll
            for (int r = 0; r < 8; ++r)
                O[(size_t)(m0 + rbase + r) * Dd + n0 + ct * 16 + nloc] = acc[ct][r];
    }
}

// ---------------------------------------------------------------------------
// Flash attention: grid = (S/128, B*H), 8 waves/WG, wave -> 16 q-rows.
// K (32x64) and V (64x32) tiles staged once per WG via async-to-LDS,
// double-buffered; all waves run the block-uniform tile count, waves past
// their causal diagonal skip compute (wave-uniform branch, EXEC stays full).
// ---------------------------------------------------------------------------
__global__ void attn_kernel(const _Float16* __restrict__ Q,
                            const _Float16* __restrict__ K,
                            const _Float16* __restrict__ Vt,
                            _Float16* __restrict__ AO) {
    __shared__ __align__(16) _Float16 Kbuf[2][32 * KPAD];
    __shared__ __align__(16) _Float16 Vbuf[2][64 * VPAD];
    __shared__ __align__(16) _Float16 plds[8][16 * 32];

    const int tid  = threadIdx.x;
    const int lane = tid & 31;
    const int wave = tid >> 5;
    const int bh   = blockIdx.y;
    const int qblk = blockIdx.x * 128;
    const int q0   = qblk + wave * 16;

    const _Float16* Qb = Q  + (size_t)bh * Ss * HDd + (size_t)q0 * HDd;
    const _Float16* Kb = K  + (size_t)bh * Ss * HDd;
    const _Float16* Vb = Vt + (size_t)bh * HDd * Ss;

    // staging roles
    const int krow = tid >> 3, kchunk = tid & 7;      // 32 rows x 8 chunks
    const int vrow = tid >> 2, vchunk = tid & 3;      // 64 rows x 4 chunks
    const _Float16* kgp = Kb + (size_t)krow * HDd + kchunk * 8;
    const _Float16* vgp = Vb + (size_t)vrow * Ss  + vchunk * 8;
    const unsigned klds[2] = {
        (unsigned)(size_t)&Kbuf[0][krow * KPAD + kchunk * 8],
        (unsigned)(size_t)&Kbuf[1][krow * KPAD + kchunk * 8] };
    const unsigned vlds[2] = {
        (unsigned)(size_t)&Vbuf[0][vrow * VPAD + vchunk * 8],
        (unsigned)(size_t)&Vbuf[1][vrow * VPAD + vchunk * 8] };

    const v16h a0 = load_a_tile(Qb, HDd, 0,  lane);   // hd 0..31
    const v16h a1 = load_a_tile(Qb, HDd, 32, lane);   // hd 32..63

    v8f acc[4] = {};
    float mrow[8], lrow[8];
#pragma unroll
    for (int r = 0; r < 8; ++r) { mrow[r] = -__builtin_inff(); lrow[r] = 0.0f; }

    const int nloc  = lane & 15;
    const int rbase = (lane >> 4) * 8;
    const int my_kt = (q0 + 15) / 32 + 1;             // this wave's tiles
    const int ktmax = qblk / 32 + 4;                  // block-uniform count

    async_b128(klds[0], kgp);                         // prologue: tile 0
    async_b128(vlds[0], vgp);

    for (int kt = 0; kt < ktmax; ++kt) {
        const int kbase = kt * 32;
        const int cur   = kt & 1;

        if (kt + 1 < ktmax) {
            async_b128(klds[cur ^ 1], kgp + (size_t)(kt + 1) * 32 * HDd);
            async_b128(vlds[cur ^ 1], vgp + (kt + 1) * 32);
            asm volatile("s_wait_asynccnt 0x2" ::: "memory");
        } else {
            asm volatile("s_wait_asynccnt 0x0" ::: "memory");
        }
        __syncthreads();                              // tile kt visible

        if (kt < my_kt) {                             // wave-uniform skip
            // scores = Q * K^T for 32 kpos
            v8f sc0 = {}, sc1 = {};
            const _Float16* kb0 = &Kbuf[cur][0];
            sc0 = wmma_f16(a0, load_b_tile(kb0,             KPAD, 0,  lane), sc0);
            sc0 = wmma_f16(a1, load_b_tile(kb0,             KPAD, 32, lane), sc0);
            sc1 = wmma_f16(a0, load_b_tile(kb0 + 16 * KPAD, KPAD, 0,  lane), sc1);
            sc1 = wmma_f16(a1, load_b_tile(kb0 + 16 * KPAD, KPAD, 32, lane), sc1);

            const bool need_mask = (kbase + 31) > q0;

            float p0[8], p1[8], tmax[8];
#pragma unroll
            for (int r = 0; r < 8; ++r) {
                float s0v = sc0[r] * SCALE;
                float s1v = sc1[r] * SCALE;
                if (need_mask) {
                    const int mg = q0 + rbase + r;
                    if (kbase + nloc      > mg) s0v = -__builtin_inff();
                    if (kbase + 16 + nloc > mg) s1v = -__builtin_inff();
                }
                p0[r] = s0v; p1[r] = s1v;
                tmax[r] = rowmax16(fmaxf(s0v, s1v));
            }

#pragma unroll
            for (int r = 0; r < 8; ++r) {
                const float mn = fmaxf(mrow[r], tmax[r]);
                const float al = __expf(mrow[r] - mn);
                mrow[r] = mn;
                const float e0 = __expf(p0[r] - mn);
                const float e1 = __expf(p1[r] - mn);
                lrow[r] = lrow[r] * al + rowsum16(e0 + e1);
#pragma unroll
                for (int ct = 0; ct < 4; ++ct) acc[ct][r] *= al;
                plds[wave][(r + rbase) * 32 + nloc]      = (_Float16)e0;
                plds[wave][(r + rbase) * 32 + 16 + nloc] = (_Float16)e1;
            }

            // same-wave LDS RAW (per-wave region): explicit DS-counter wait
            asm volatile("s_wait_dscnt 0x0" ::: "memory");

            const v16h pa = load_a_tile(&plds[wave][0], 32, 0, lane);
#pragma unroll
            for (int ct = 0; ct < 4; ++ct) {
                v16h vb = load_b_tile(&Vbuf[cur][ct * 16 * VPAD], VPAD, 0, lane);
                acc[ct] = wmma_f16(pa, vb, acc[ct]);
            }
        }
        __syncthreads();                              // reads done before reuse
    }

    // normalize, store f16 into [B, S, H*64] for the Wo projection
    const int b = bh >> 4, h = bh & 15;
#pragma unroll
    for (int r = 0; r < 8; ++r) {
        const float rinv = 1.0f / lrow[r];
        const int s = q0 + rbase + r;
        const size_t rowoff = ((size_t)(b * Ss + s)) * Dd + h * HDd;
#pragma unroll
        for (int ct = 0; ct < 4; ++ct)
            AO[rowoff + ct * 16 + nloc] = (_Float16)(acc[ct][r] * rinv);
    }
}

// ---------------------------------------------------------------------------
extern "C" void kernel_launch(void* const* d_in, const int* in_sizes, int n_in,
                              void* d_out, int out_size, void* d_ws, size_t ws_size,
                              hipStream_t stream) {
    const float* x  = (const float*)d_in[0];
    const float* Wq = (const float*)d_in[1];
    const float* Wk = (const float*)d_in[2];
    const float* Wv = (const float*)d_in[3];
    const float* Wo = (const float*)d_in[4];
    (void)in_sizes; (void)n_in; (void)out_size; (void)ws_size;

    char* ws = (char*)d_ws;
    size_t off = 0;
    auto carve = [&](size_t halves) {
        _Float16* p = (_Float16*)(ws + off);
        off += halves * sizeof(_Float16);
        return p;
    };
    _Float16* Xh  = carve((size_t)Mrows * Dd);
    _Float16* Wqh = carve((size_t)Dd * Dd);
    _Float16* Wkh = carve((size_t)Dd * Dd);
    _Float16* Wvh = carve((size_t)Dd * Dd);
    _Float16* Woh = carve((size_t)Dd * Dd);
    _Float16* Qb  = carve((size_t)Mrows * Dd);   // [B,H,S,64]
    _Float16* Kb  = carve((size_t)Mrows * Dd);   // [B,H,S,64]
    _Float16* Vt  = carve((size_t)Mrows * Dd);   // [B,H,64,S]
    _Float16* AO  = carve((size_t)Mrows * Dd);   // [B,S,1024]

    cvt_f32_f16_kernel<<<2048, 256, 0, stream>>>(x,  Xh,  Mrows * Dd / 4);
    cvt_f32_f16_kernel<<<1024, 256, 0, stream>>>(Wq, Wqh, Dd * Dd / 4);
    cvt_f32_f16_kernel<<<1024, 256, 0, stream>>>(Wk, Wkh, Dd * Dd / 4);
    cvt_f32_f16_kernel<<<1024, 256, 0, stream>>>(Wv, Wvh, Dd * Dd / 4);
    cvt_f32_f16_kernel<<<1024, 256, 0, stream>>>(Wo, Woh, Dd * Dd / 4);

    dim3 gg(32, 16);
    gemm_kernel<<<gg, 256, 0, stream>>>(Xh, Wqh, (void*)Qb, 0);  // Q + RoPE
    gemm_kernel<<<gg, 256, 0, stream>>>(Xh, Wkh, (void*)Kb, 0);  // K + RoPE
    gemm_kernel<<<gg, 256, 0, stream>>>(Xh, Wvh, (void*)Vt, 1);  // V transposed

    attn_kernel<<<dim3(Ss / 128, Bb * Hh), 256, 0, stream>>>(Qb, Kb, Vt, AO);

    gemm_kernel<<<gg, 256, 0, stream>>>(AO, Woh, d_out, 2);      // out = AO*Wo^T
}
